// AttentionBlock_32736240731028
// MI455X (gfx1250) — matrix operand order repaired and checked by reading, not compiled, not run on hardware
//
#include <hip/hip_runtime.h>
#include <hip/hip_bf16.h>

#define BB 4
#define CCH 512
#define GRP 32
#define CGRP 16
#define HW 4096
#define EPSV 1e-5f

typedef __attribute__((ext_vector_type(16))) __bf16 v16bf;
typedef __attribute__((ext_vector_type(16))) unsigned short v16u;
typedef __attribute__((ext_vector_type(8)))  float v8f;

__device__ __forceinline__ unsigned short f2bf(float f) {
  unsigned int u = __float_as_uint(f);
  u += 0x7FFFu + ((u >> 16) & 1u);           // round-to-nearest-even
  return (unsigned short)(u >> 16);
}

__device__ __forceinline__ v8f v8f_zero() {
  v8f z;
#pragma unroll
  for (int i = 0; i < 8; ++i) z[i] = 0.0f;
  return z;
}

__device__ __forceinline__ v8f wmma_bf16(v16u a, v16u b, v8f c) {
  return __builtin_amdgcn_wmma_f32_16x16x32_bf16(
      false, __builtin_bit_cast(v16bf, a),
      false, __builtin_bit_cast(v16bf, b),
      (short)0, c, false, false);
}

// Async Global->LDS copy (CDNA5): 16 bytes per lane, tracked by ASYNCcnt.
// LDS dest address = low 32 bits of the generic pointer (LDS segment offset).
__device__ __forceinline__ void async_copy_b128(void* lds_dst, const void* gsrc) {
  unsigned lds = (unsigned)(uintptr_t)lds_dst;
  asm volatile("global_load_async_to_lds_b128 %0, %1, off"
               :: "v"(lds), "v"(gsrc)
               : "memory");
}
__device__ __forceinline__ void wait_async0() {
  asm volatile("s_wait_asynccnt 0x0" ::: "memory");
}

// ---------------------------------------------------------------------------
// Kernel 1: GroupNorm (32 groups of 16 channels) -> bf16 activations
// ---------------------------------------------------------------------------
__global__ __launch_bounds__(256)
void gn_kernel(const float* __restrict__ x, const float* __restrict__ gamma,
               const float* __restrict__ beta, unsigned short* __restrict__ hn) {
  const int bg = blockIdx.x;
  const int b = bg >> 5, g = bg & 31;
  const int t = threadIdx.x;
  const int c = t & 15, prow = t >> 4;
  const float* xb = x + (size_t)b * HW * CCH + g * CGRP + c;
  float s = 0.f, ss = 0.f;
  for (int p = prow; p < HW; p += 16) {
    float v = xb[(size_t)p * CCH];
    s += v; ss += v * v;
  }
  __shared__ float red[2][256];
  red[0][t] = s; red[1][t] = ss;
  __syncthreads();
  for (int off = 128; off > 0; off >>= 1) {
    if (t < off) { red[0][t] += red[0][t + off]; red[1][t] += red[1][t + off]; }
    __syncthreads();
  }
  const float inv_n = 1.0f / (HW * CGRP);
  float mean = red[0][0] * inv_n;
  float var  = red[1][0] * inv_n - mean * mean;
  float rstd = rsqrtf(var + EPSV);
  float gm = gamma[g * CGRP + c], bt = beta[g * CGRP + c];
  unsigned short* ob = hn + (size_t)b * HW * CCH + g * CGRP + c;
  for (int p = prow; p < HW; p += 16) {
    float v = xb[(size_t)p * CCH];
    ob[(size_t)p * CCH] = f2bf((v - mean) * rstd * gm + bt);
  }
}

// ---------------------------------------------------------------------------
// Kernel 2: bf16 WMMA GEMM  OUT[M,512] = (A[M,512]*W[512,512] + bias)*scale
// MODE 0: bf16 row-major out; MODE 1: bf16 out transposed per batch
// ([b][c][hw], for V); MODE 2: f32 out + residual.
// Block tile 64x128, 256 threads (8 waves, 2x4), each wave 32x32 = 2x2 WMMA.
// A tile staged via async-to-LDS; W tile converted f32->bf16 and stored
// TRANSPOSED so B-fragments load as ds_load_b128.
// ---------------------------------------------------------------------------
template <int MODE>
__global__ __launch_bounds__(256)
void gemm_kernel(const unsigned short* __restrict__ A,
                 const float* __restrict__ W,
                 const float* __restrict__ bias,
                 const float* __restrict__ resid,
                 float scale,
                 unsigned short* __restrict__ out_bf,
                 float* __restrict__ out_f) {
  const int rb = blockIdx.x * 64;
  const int cb = blockIdx.y * 128;
  const int t = threadIdx.x;
  const int w = t >> 5, l = t & 31, ll = l & 15, lh = l >> 4;
  const int wm = w & 1, wn = w >> 1;

  __shared__ unsigned short As[64 * 40];     // 64 rows x 32 k   (stride 40)
  __shared__ unsigned short WsT[128 * 40];   // 128 cols x 32 k  (stride 40)

  v8f acc[2][2];
#pragma unroll
  for (int i = 0; i < 2; ++i)
#pragma unroll
    for (int j = 0; j < 2; ++j) acc[i][j] = v8f_zero();

  // per-thread A-tile chunk: 256 b128 chunks cover 64x32 bf16
  const int ar = t >> 2, acol = (t & 3) * 8;

  for (int k0 = 0; k0 < CCH; k0 += 32) {
    // A tile: one async b128 per thread (no VGPR staging)
    async_copy_b128(&As[ar * 40 + acol],
                    &A[(size_t)(rb + ar) * CCH + k0 + acol]);
    // W tile: load row-major float4, store transposed bf16 (col-major)
#pragma unroll
    for (int i = 0; i < 4; ++i) {
      int ch = t + i * 256;
      int r = ch >> 5, cc = (ch & 31) * 4;
      float4 wv = *(const float4*)&W[(size_t)(k0 + r) * CCH + cb + cc];
      WsT[(cc + 0) * 40 + r] = f2bf(wv.x);
      WsT[(cc + 1) * 40 + r] = f2bf(wv.y);
      WsT[(cc + 2) * 40 + r] = f2bf(wv.z);
      WsT[(cc + 3) * 40 + r] = f2bf(wv.w);
    }
    wait_async0();
    __syncthreads();

    v16u af[2], bfr[2];
#pragma unroll
    for (int i = 0; i < 2; ++i) {
      int row = wm * 32 + i * 16 + ll;
#pragma unroll
      for (int e = 0; e < 16; ++e)
        af[i][e] = As[row * 40 + lh * 16 + e];
      int col = wn * 32 + i * 16 + ll;
#pragma unroll
      for (int e = 0; e < 16; ++e)
        bfr[i][e] = WsT[col * 40 + lh * 16 + e];
    }
#pragma unroll
    for (int i = 0; i < 2; ++i)
#pragma unroll
      for (int j = 0; j < 2; ++j)
        acc[i][j] = wmma_bf16(af[i], bfr[j], acc[i][j]);
    __syncthreads();
  }

#pragma unroll
  for (int i = 0; i < 2; ++i)
#pragma unroll
    for (int j = 0; j < 2; ++j) {
      int col = cb + wn * 32 + j * 16 + ll;
      float bv = bias[col];
#pragma unroll
      for (int r8 = 0; r8 < 8; ++r8) {
        int row = rb + wm * 32 + i * 16 + lh * 8 + r8;
        float v = (acc[i][j][r8] + bv) * scale;
        if constexpr (MODE == 2) {
          v += resid[(size_t)row * CCH + col];
          out_f[(size_t)row * CCH + col] = v;
        } else if constexpr (MODE == 1) {
          // [b][c][hw] layout: b = row>>12, hw = row&4095
          out_bf[((size_t)(row >> 12) * CCH + col) * HW + (row & 4095)] = f2bf(v);
        } else {
          out_bf[(size_t)row * CCH + col] = f2bf(v);
        }
      }
    }
}

// ---------------------------------------------------------------------------
// Kernel 3: flash attention, seq 4096, d=512 (scale pre-folded into Q).
// grid (HW/64, B), 128 threads = 4 waves; each wave owns 16 query rows.
// KV tile = 32 keys. K tile staged key-major [32][520]; V tile staged
// channel-major [512][40] from the pre-transposed V^T so both S and PV
// B-fragments are contiguous ds_load_b128 pairs. K/V share one LDS buffer;
// both tiles are staged with async Global->LDS copies (ASYNCcnt).
// ---------------------------------------------------------------------------
__global__ __launch_bounds__(128)
void flash_kernel(const unsigned short* __restrict__ Q,
                  const unsigned short* __restrict__ Km,
                  const unsigned short* __restrict__ Vt,   // [B][C][HW]
                  unsigned short* __restrict__ O) {
  const int b = blockIdx.y;
  const int qbase = blockIdx.x * 64;
  const int t = threadIdx.x;
  const int w = t >> 5, l = t & 31, ll = l & 15, lh = l >> 4;

  __shared__ unsigned short KVs[512 * 40];   // K phase: [32][520]; V: [512][40]
  __shared__ unsigned short Ps[4][16][40];   // per-wave P staging (16x32)

  // Preload Q as 16 A-fragments (16x32 each) covering d=512.
  const unsigned short* Qrow = Q + ((size_t)b * HW + qbase + w * 16 + ll) * CCH;
  v16u qa[16];
#pragma unroll
  for (int f = 0; f < 16; ++f)
#pragma unroll
    for (int e = 0; e < 16; ++e)
      qa[f][e] = Qrow[f * 32 + lh * 16 + e];

  v8f oacc[32];
#pragma unroll
  for (int i = 0; i < 32; ++i) oacc[i] = v8f_zero();
  float mrow[8], lrow[8];
#pragma unroll
  for (int j = 0; j < 8; ++j) { mrow[j] = -3.0e38f; lrow[j] = 0.f; }

  for (int jb = 0; jb < HW; jb += 32) {
    // ---- async-load K tile (key-major, rows of 512 bf16, stride 520) ----
    const unsigned short* Kb = Km + ((size_t)b * HW + jb) * CCH;
#pragma unroll
    for (int i = 0; i < 16; ++i) {
      int ch = t + i * 128;
      int r = ch >> 6, cc = (ch & 63) * 8;
      async_copy_b128(&KVs[r * 520 + cc], &Kb[r * CCH + cc]);
    }
    wait_async0();
    __syncthreads();

    // ---- S = Q * K^T (two 16x16 accumulators over 32 keys) ----
    v8f s0 = v8f_zero(), s1 = v8f_zero();
#pragma unroll
    for (int f = 0; f < 16; ++f) {
      v16u b0, b1;
#pragma unroll
      for (int e = 0; e < 16; ++e) {
        int kc = f * 32 + lh * 16 + e;
        b0[e] = KVs[ll * 520 + kc];
        b1[e] = KVs[(ll + 16) * 520 + kc];
      }
      s0 = wmma_bf16(qa[f], b0, s0);
      s1 = wmma_bf16(qa[f], b1, s1);
    }

    // ---- online softmax (row r = lh*8 + j lives in vgpr j of a lane half) ----
    float alpha[8];
#pragma unroll
    for (int j = 0; j < 8; ++j) {
      float v = fmaxf(s0[j], s1[j]);
      v = fmaxf(v, __shfl_xor(v, 1, 32));
      v = fmaxf(v, __shfl_xor(v, 2, 32));
      v = fmaxf(v, __shfl_xor(v, 4, 32));
      v = fmaxf(v, __shfl_xor(v, 8, 32));
      float mn = fmaxf(mrow[j], v);
      alpha[j] = __expf(mrow[j] - mn);
      mrow[j] = mn;
      float p0 = __expf(s0[j] - mn);
      float p1 = __expf(s1[j] - mn);
      s0[j] = p0; s1[j] = p1;
      float rs = p0 + p1;
      rs += __shfl_xor(rs, 1, 32);
      rs += __shfl_xor(rs, 2, 32);
      rs += __shfl_xor(rs, 4, 32);
      rs += __shfl_xor(rs, 8, 32);
      lrow[j] = lrow[j] * alpha[j] + rs;
    }
#pragma unroll
    for (int i = 0; i < 32; ++i)
#pragma unroll
      for (int j = 0; j < 8; ++j) oacc[i][j] *= alpha[j];

    // stage P (C-layout -> LDS) for A-layout reload
#pragma unroll
    for (int j = 0; j < 8; ++j) {
      Ps[w][lh * 8 + j][ll]      = f2bf(s0[j]);
      Ps[w][lh * 8 + j][16 + ll] = f2bf(s1[j]);
    }
    __syncthreads();   // K reads + P writes complete

    // ---- async-load V tile channel-major from V^T: 512 x 32, stride 40 ----
    const unsigned short* Vb = Vt + (size_t)b * CCH * HW + jb;
#pragma unroll
    for (int i = 0; i < 16; ++i) {
      int ch = t + i * 128;
      int r = ch >> 2, kk = (ch & 3) * 8;
      async_copy_b128(&KVs[r * 40 + kk], &Vb[(size_t)r * HW + kk]);
    }
    // overlap: reload P fragment while the copy engine fills the V tile
    v16u pa;
#pragma unroll
    for (int e = 0; e < 16; ++e)
      pa[e] = Ps[w][ll][lh * 16 + e];
    wait_async0();
    __syncthreads();   // V resident

    // ---- O += P * V across all 32 channel tiles (contiguous b128 frags) ----
#pragma unroll
    for (int tt = 0; tt < 32; ++tt) {
      v16u bv;
#pragma unroll
      for (int e = 0; e < 16; ++e)
        bv[e] = KVs[(tt * 16 + ll) * 40 + lh * 16 + e];
      oacc[tt] = wmma_bf16(pa, bv, oacc[tt]);
    }
    __syncthreads();   // V reads done before next K load
  }

  float rl[8];
#pragma unroll
  for (int j = 0; j < 8; ++j) rl[j] = 1.0f / lrow[j];
  unsigned short* Ob = O + ((size_t)b * HW + qbase + w * 16) * CCH;
#pragma unroll
  for (int tt = 0; tt < 32; ++tt)
#pragma unroll
    for (int j = 0; j < 8; ++j)
      Ob[(size_t)(lh * 8 + j) * CCH + tt * 16 + ll] = f2bf(oacc[tt][j] * rl[j]);
}

// ---------------------------------------------------------------------------
extern "C" void kernel_launch(void* const* d_in, const int* in_sizes, int n_in,
                              void* d_out, int out_size, void* d_ws, size_t ws_size,
                              hipStream_t stream) {
  const float* x     = (const float*)d_in[0];
  const float* gamma = (const float*)d_in[1];
  const float* beta  = (const float*)d_in[2];
  const float* wq    = (const float*)d_in[3];
  const float* bq    = (const float*)d_in[4];
  const float* wk    = (const float*)d_in[5];
  const float* bk    = (const float*)d_in[6];
  const float* wv    = (const float*)d_in[7];
  const float* bv    = (const float*)d_in[8];
  const float* wp    = (const float*)d_in[9];
  const float* bp    = (const float*)d_in[10];
  float* out = (float*)d_out;

  char* ws = (char*)d_ws;
  const size_t buf = (size_t)BB * HW * CCH * sizeof(unsigned short); // 16 MB
  unsigned short* hn = (unsigned short*)(ws + 0 * buf);
  unsigned short* q  = (unsigned short*)(ws + 1 * buf);
  unsigned short* k  = (unsigned short*)(ws + 2 * buf);
  unsigned short* vt = (unsigned short*)(ws + 3 * buf);  // V^T [B][C][HW]
  unsigned short* ao = (unsigned short*)(ws + 4 * buf);

  gn_kernel<<<BB * GRP, 256, 0, stream>>>(x, gamma, beta, hn);

  dim3 gg(BB * HW / 64, CCH / 128);
  const float scale = 0.04419417382415922f;  // 1/sqrt(512)
  gemm_kernel<0><<<gg, 256, 0, stream>>>(hn, wq, bq, nullptr, scale, q, nullptr);
  gemm_kernel<0><<<gg, 256, 0, stream>>>(hn, wk, bk, nullptr, 1.0f, k, nullptr);
  gemm_kernel<1><<<gg, 256, 0, stream>>>(hn, wv, bv, nullptr, 1.0f, vt, nullptr);

  flash_kernel<<<dim3(HW / 64, BB), 128, 0, stream>>>(q, k, vt, ao);

  gemm_kernel<2><<<gg, 256, 0, stream>>>(ao, wp, bp, x, 1.0f, nullptr, out);
}
